// MultiHeadAttention_33904471834906
// MI455X (gfx1250) — matrix operand order repaired and checked by reading, not compiled, not run on hardware
//
#include <hip/hip_runtime.h>

// ---------- types ----------
typedef __attribute__((ext_vector_type(16))) __bf16 v16bf;
typedef __attribute__((ext_vector_type(8)))  __bf16 v8bf;
typedef __attribute__((ext_vector_type(8)))  float  v8f;
typedef __attribute__((ext_vector_type(4)))  float  v4f;

#define NB 4
#define SS 1024
#define DM 1024
#define NH 16
#define HD 64

__device__ __forceinline__ __bf16 f2bf(float x) { return (__bf16)x; }  // native cvt

__device__ __forceinline__ v8f wmma_bf16(v16bf a, v16bf b, v8f c) {
  return __builtin_amdgcn_wmma_f32_16x16x32_bf16(false, a, false, b,
                                                 (short)0, c, false, false);
}

// A fragment (16x32, lane%16 = M, K contiguous 16 per half-wave group)
__device__ __forceinline__ v16bf loadA_f32(const float* __restrict__ xr, int kk, int grp) {
  const v4f* p0 = (const v4f*)(xr + kk + 16 * grp);
  const v4f* p1 = (const v4f*)(xr + kk + 16 * grp + 8);
  v4f f0 = p0[0], f1 = p0[1], f2 = p1[0], f3 = p1[1];
  v16bf a;
#pragma unroll
  for (int i = 0; i < 4; ++i) {
    a[i]      = f2bf(f0[i]);  a[4 + i]  = f2bf(f1[i]);
    a[8 + i]  = f2bf(f2[i]);  a[12 + i] = f2bf(f3[i]);
  }
  return a;
}
__device__ __forceinline__ v16bf loadA_bf16(const __bf16* __restrict__ xr, int kk, int grp) {
  v8bf lo = *(const v8bf*)(xr + kk + 16 * grp);
  v8bf hi = *(const v8bf*)(xr + kk + 16 * grp + 8);
  v16bf a;
#pragma unroll
  for (int i = 0; i < 8; ++i) { a[i] = lo[i]; a[8 + i] = hi[i]; }
  return a;
}

// ---------- weight transpose + bf16 convert (coalesced via 32x32 LDS tile) ----------
__global__ __launch_bounds__(256) void wt_kernel(const float* __restrict__ W,
                                                 __bf16* __restrict__ Wt) {
  __shared__ float tile[32][33];
  const int tx = threadIdx.x & 31, ty = threadIdx.x >> 5;   // 32 x 8
  const int n0 = (blockIdx.x & 31) * 32, k0 = (blockIdx.x >> 5) * 32;
#pragma unroll
  for (int j = 0; j < 32; j += 8)
    tile[ty + j][tx] = W[(k0 + ty + j) * DM + n0 + tx];
  __syncthreads();
#pragma unroll
  for (int j = 0; j < 32; j += 8)
    Wt[(n0 + ty + j) * DM + k0 + tx] = f2bf(tile[tx][ty + j]);
}

// ---------- GEMM: C[4096x1024] = X @ W + b ; per-wave 32x64 strip ----------
// Ping-pong K pipeline: two register sets, peeled epilogue, no buffer copies.
// MODE 0: Q bf16 [B,H,S,64]   MODE 1: K bf16 [B,H,S,64]
// MODE 2: Vt bf16 [B,H,64,S]  MODE 3: A bf16 ctx, out fp32 [B,S,D] (NT stores)
template <int MODE>
__global__ __launch_bounds__(256) void gemm_kernel(const float*  __restrict__ Xf,
                                                   const __bf16* __restrict__ Xbf,
                                                   const __bf16* __restrict__ Wt,
                                                   const float*  __restrict__ bias,
                                                   __bf16* __restrict__ outBf,
                                                   float*  __restrict__ outF) {
  const int lane  = threadIdx.x & 31;
  const int wave  = threadIdx.x >> 5;
  const int w     = blockIdx.x * 8 + wave;     // 2048 waves
  const int mtile = w >> 4;                    // 0..127 (32 rows each)
  const int nt64  = w & 15;                    // 0..15  (64 cols each)
  const int m     = lane & 15;
  const int grp   = lane >> 4;
  const int row0  = mtile * 32 + m;
  const int row1  = row0 + 16;
  const __bf16* wbase = Wt + (nt64 * 64 + m) * DM + 16 * grp;

  auto ldA = [&](int kk, int which) -> v16bf {
    if constexpr (MODE < 3)
      return loadA_f32(Xf + (which ? row1 : row0) * DM, kk, grp);
    else
      return loadA_bf16(Xbf + (which ? row1 : row0) * DM, kk, grp);
  };
  auto ldB = [&](int nt, int kk) -> v16bf {
    return *(const v16bf*)(wbase + nt * 16 * DM + kk);
  };

  v8f acc[2][4] = {};
  v16bf a0[2], a1[2], b0[4], b1[4];

  // prologue: chunk 0 -> set 0
  a0[0] = ldA(0, 0);  a0[1] = ldA(0, 1);
#pragma unroll
  for (int nt = 0; nt < 4; ++nt) b0[nt] = ldB(nt, 0);

  for (int kk = 0; kk < DM - 64; kk += 64) {
    // prefetch kk+32 -> set 1
    a1[0] = ldA(kk + 32, 0);  a1[1] = ldA(kk + 32, 1);
#pragma unroll
    for (int nt = 0; nt < 4; ++nt) b1[nt] = ldB(nt, kk + 32);
#pragma unroll
    for (int nt = 0; nt < 4; ++nt) {
      acc[0][nt] = wmma_bf16(a0[0], b0[nt], acc[0][nt]);
      acc[1][nt] = wmma_bf16(a0[1], b0[nt], acc[1][nt]);
    }
    // prefetch kk+64 -> set 0
    a0[0] = ldA(kk + 64, 0);  a0[1] = ldA(kk + 64, 1);
#pragma unroll
    for (int nt = 0; nt < 4; ++nt) b0[nt] = ldB(nt, kk + 64);
#pragma unroll
    for (int nt = 0; nt < 4; ++nt) {
      acc[0][nt] = wmma_bf16(a1[0], b1[nt], acc[0][nt]);
      acc[1][nt] = wmma_bf16(a1[1], b1[nt], acc[1][nt]);
    }
  }
  // epilogue: chunks DM-64 (set 0) and DM-32 (set 1)
  a1[0] = ldA(DM - 32, 0);  a1[1] = ldA(DM - 32, 1);
#pragma unroll
  for (int nt = 0; nt < 4; ++nt) b1[nt] = ldB(nt, DM - 32);
#pragma unroll
  for (int nt = 0; nt < 4; ++nt) {
    acc[0][nt] = wmma_bf16(a0[0], b0[nt], acc[0][nt]);
    acc[1][nt] = wmma_bf16(a0[1], b0[nt], acc[1][nt]);
  }
#pragma unroll
  for (int nt = 0; nt < 4; ++nt) {
    acc[0][nt] = wmma_bf16(a1[0], b1[nt], acc[0][nt]);
    acc[1][nt] = wmma_bf16(a1[1], b1[nt], acc[1][nt]);
  }

#pragma unroll
  for (int nt = 0; nt < 4; ++nt) {
    const int ncol = nt64 * 64 + nt * 16 + m;
    const float bv = bias[ncol];
#pragma unroll
    for (int mt = 0; mt < 2; ++mt) {
#pragma unroll
      for (int r = 0; r < 8; ++r) {
        const int seq = mtile * 32 + mt * 16 + r + 8 * grp;
        const float v = acc[mt][nt][r] + bv;
        if constexpr (MODE == 3) {
          __builtin_nontemporal_store(v, outF + seq * DM + ncol);
        } else {
          const int b = seq >> 10, s = seq & 1023;
          const int h = ncol >> 6, d = ncol & 63;
          if constexpr (MODE == 2)
            outBf[((b * NH + h) * HD + d) * SS + s] = f2bf(v);   // Vt [B,H,64,S]
          else
            outBf[((b * NH + h) * SS + s) * HD + d] = f2bf(v);   // Q/K [B,H,S,64]
        }
      }
    }
  }
}

// ---------- fused attention ----------
#define SC_STRIDE 1032   // floats: pad vs 64-bank LDS
#define AB_STRIDE 1040   // bf16 halves

__global__ __launch_bounds__(128) void attn_kernel(const __bf16* __restrict__ Qbf,
                                                   const __bf16* __restrict__ Kbf,
                                                   const __bf16* __restrict__ Vt,
                                                   const int*    __restrict__ mask,
                                                   float*  __restrict__ attnOut,
                                                   __bf16* __restrict__ ctxBf) {
  __shared__ __align__(32) float  sc[16 * SC_STRIDE];
  __shared__ __align__(32) __bf16 ab[16 * AB_STRIDE];

  const int lane = threadIdx.x & 31;
  const int wave = threadIdx.x >> 5;             // 4 waves
  const int bid  = blockIdx.x;                   // 4096 = B*H*(S/16)
  const int qb   = bid & 63;
  const int h    = (bid >> 6) & 15;
  const int b    = bid >> 10;
  const int q0   = qb * 16;
  const int m    = lane & 15;
  const int grp  = lane >> 4;

  const __bf16* qhead = Qbf + (b * NH + h) * SS * HD;
  const __bf16* khead = Kbf + (b * NH + h) * SS * HD;
  const __bf16* vhead = Vt  + (b * NH + h) * HD * SS;

  // Q block A-fragments (hd=64 -> two K=32 chunks), reused for all key tiles
  v16bf aq0, aq1;
  {
    const __bf16* qr = qhead + (q0 + m) * HD;
    aq0 = loadA_bf16(qr, 0, grp);
    aq1 = loadA_bf16(qr, 32, grp);
  }

  // ---- pass 1: scores into LDS (scaled); fully unrolled (16 tiles/wave) ----
  const __bf16* kbase = khead + m * HD + 16 * grp;
#pragma unroll
  for (int i = 0; i < 16; ++i) {
    const int kt = wave + 4 * i;
    const __bf16* kp = kbase + kt * (16 * HD);
    v16bf bm0 = *(const v16bf*)(kp);
    v16bf bm1 = *(const v16bf*)(kp + 32);
    v8f c8 = {};
    c8 = wmma_bf16(aq0, bm0, c8);
    c8 = wmma_bf16(aq1, bm1, c8);
    const int k0 = kt * 16;
#pragma unroll
    for (int r = 0; r < 8; ++r)
      sc[(r + 8 * grp) * SC_STRIDE + k0 + m] = c8[r] * 0.125f;
  }
  __syncthreads();

  // ---- pass 2: masked softmax; write attn (fp32, NT) + bf16 copy in LDS ----
  const float NEG_INF = -__builtin_inff();
  for (int rr = 0; rr < 4; ++rr) {
    const int rowq = wave * 4 + rr;
    const int* mrow = mask + ((long)b * SS + q0 + rowq) * SS;
    float* srow = sc + rowq * SC_STRIDE;

    float mx = NEG_INF;
    for (int c = lane; c < SS; c += 32) {
      float s = srow[c];
      if (__builtin_nontemporal_load(mrow + c) == 0) s = NEG_INF;
      srow[c] = s;
      mx = fmaxf(mx, s);
    }
#pragma unroll
    for (int off = 16; off > 0; off >>= 1) mx = fmaxf(mx, __shfl_xor(mx, off, 32));

    float sum = 0.f;
    for (int c = lane; c < SS; c += 32) {
      float e = __expf(srow[c] - mx);
      srow[c] = e;
      sum += e;
    }
#pragma unroll
    for (int off = 16; off > 0; off >>= 1) sum += __shfl_xor(sum, off, 32);
    const float inv = 1.0f / sum;

    float* aout = attnOut + (((long)(b * NH + h) * SS) + q0 + rowq) * SS;
    __bf16* arow = ab + rowq * AB_STRIDE;
    for (int c = lane; c < SS; c += 32) {
      float p = srow[c] * inv;
      __builtin_nontemporal_store(p, aout + c);   // 268MB stream: keep out of L2
      arow[c] = f2bf(p);
    }
  }
  __syncthreads();

  // ---- pass 3: ctx = attn @ V ; ping-pong V tiles, peeled epilogue ----
  v8f c8 = {};
  const __bf16* arow = ab + m * AB_STRIDE;
  const __bf16* vrow = vhead + (wave * 16 + m) * SS + 16 * grp;
  v16bf vb0 = *(const v16bf*)(vrow);
  v16bf vb1;
  for (int kk = 0; kk < SS - 64; kk += 64) {
    vb1 = *(const v16bf*)(vrow + kk + 32);
    c8  = wmma_bf16(loadA_bf16(arow, kk, grp), vb0, c8);
    vb0 = *(const v16bf*)(vrow + kk + 64);
    c8  = wmma_bf16(loadA_bf16(arow, kk + 32, grp), vb1, c8);
  }
  vb1 = *(const v16bf*)(vrow + SS - 32);
  c8  = wmma_bf16(loadA_bf16(arow, SS - 64, grp), vb0, c8);
  c8  = wmma_bf16(loadA_bf16(arow, SS - 32, grp), vb1, c8);

#pragma unroll
  for (int r = 0; r < 8; ++r) {
    const int seq = q0 + r + 8 * grp;
    const int dm  = h * HD + wave * 16 + m;
    ctxBf[((long)b * SS + seq) * DM + dm] = f2bf(c8[r]);
  }
}

// ---------- launcher ----------
extern "C" void kernel_launch(void* const* d_in, const int* in_sizes, int n_in,
                              void* d_out, int out_size, void* d_ws, size_t ws_size,
                              hipStream_t stream) {
  const float* query = (const float*)d_in[0];
  const float* key   = (const float*)d_in[1];
  const float* value = (const float*)d_in[2];
  const int*   mask  = (const int*)  d_in[3];
  const float* Wq = (const float*)d_in[4];  const float* bq = (const float*)d_in[5];
  const float* Wk = (const float*)d_in[6];  const float* bk = (const float*)d_in[7];
  const float* Wv = (const float*)d_in[8];  const float* bv = (const float*)d_in[9];
  const float* Wo = (const float*)d_in[10]; const float* bo = (const float*)d_in[11];

  char* ws = (char*)d_ws;
  const size_t MB = 1024ull * 1024ull;
  __bf16* WqT  = (__bf16*)(ws + 0 * MB);
  __bf16* WkT  = (__bf16*)(ws + 2 * MB);
  __bf16* WvT  = (__bf16*)(ws + 4 * MB);
  __bf16* WoT  = (__bf16*)(ws + 6 * MB);
  __bf16* Qbf  = (__bf16*)(ws + 8 * MB);    // [B,H,S,64]
  __bf16* Kbf  = (__bf16*)(ws + 16 * MB);   // [B,H,S,64]
  __bf16* Vtbf = (__bf16*)(ws + 24 * MB);   // [B,H,64,S]
  __bf16* Ctx  = (__bf16*)(ws + 32 * MB);   // [B,S,D]

  float* out     = (float*)d_out;
  float* attnOut = out + (size_t)NB * SS * DM;   // outputs concatenated: out, attn

  // 1) weight transposes (fp32 -> bf16, coalesced)
  wt_kernel<<<1024, 256, 0, stream>>>(Wq, WqT);
  wt_kernel<<<1024, 256, 0, stream>>>(Wk, WkT);
  wt_kernel<<<1024, 256, 0, stream>>>(Wv, WvT);
  wt_kernel<<<1024, 256, 0, stream>>>(Wo, WoT);

  // 2) projections (2048 waves each)
  gemm_kernel<0><<<256, 256, 0, stream>>>(query, nullptr, WqT, bq, Qbf,  nullptr);
  gemm_kernel<1><<<256, 256, 0, stream>>>(key,   nullptr, WkT, bk, Kbf,  nullptr);
  gemm_kernel<2><<<256, 256, 0, stream>>>(value, nullptr, WvT, bv, Vtbf, nullptr);

  // 3) fused attention (scores + masked softmax + attn output + ctx)
  attn_kernel<<<NB * NH * (SS / 16), 128, 0, stream>>>(Qbf, Kbf, Vtbf, mask,
                                                       attnOut, Ctx);

  // 4) output projection (fp32 result, NT stores)
  gemm_kernel<3><<<256, 256, 0, stream>>>(nullptr, Ctx, WoT, bo, nullptr, out);
}
